// ChatSARAN_MLP_22522808500593
// MI455X (gfx1250) — matrix operand order repaired and checked by reading, not compiled, not run on hardware
//
#include <hip/hip_runtime.h>
#include <math.h>

typedef __bf16 bf16;
typedef __attribute__((ext_vector_type(16))) __bf16 v16bf;
typedef __attribute__((ext_vector_type(4)))  __bf16 v4bf;
typedef __attribute__((ext_vector_type(8)))  float  v8f;
typedef __attribute__((ext_vector_type(4)))  unsigned int u32x4;

#define TSEQ  2048
#define DMODEL 1024
#define NVOCAB 32000
#define NBATCH 2

union Frag16 { v16bf v; u32x4 q[2]; };

// ---------------------------------------------------------------------------
// Embedding: x[b,t,:] = tok_emb[idx[b,t],:] + pos_emb[t,:]   (f32 + bf16 copy)
// grid = B*T blocks, 256 threads, 4 floats/thread
// ---------------------------------------------------------------------------
__global__ __launch_bounds__(256) void embed_kernel(
    const int* __restrict__ idx, const float* __restrict__ tok,
    const float* __restrict__ pos, float* __restrict__ x, bf16* __restrict__ xh)
{
    int bt = blockIdx.x;
    int t  = bt & (TSEQ - 1);
    int id = idx[bt];
    int d  = threadIdx.x * 4;
    float4 a = *(const float4*)(tok + (long long)id * DMODEL + d);
    float4 b = *(const float4*)(pos + (long long)t  * DMODEL + d);
    float4 r; r.x = a.x + b.x; r.y = a.y + b.y; r.z = a.z + b.z; r.w = a.w + b.w;
    *(float4*)(x + (long long)bt * DMODEL + d) = r;
    v4bf h; h.x = (bf16)r.x; h.y = (bf16)r.y; h.z = (bf16)r.z; h.w = (bf16)r.w;
    *(v4bf*)(xh + (long long)bt * DMODEL + d) = h;
}

// ---------------------------------------------------------------------------
// Transpose-convert f32 [R][C] -> bf16 [C][R]   (weights -> Wt[n][k])
// block (32,8), tile 32x32 via LDS
// ---------------------------------------------------------------------------
__global__ __launch_bounds__(256) void wt_f32_to_bf16_t(
    const float* __restrict__ in, bf16* __restrict__ out, int R, int C)
{
    __shared__ float tile[32][33];
    int r0 = blockIdx.y * 32, c0 = blockIdx.x * 32;
    #pragma unroll
    for (int i = threadIdx.y; i < 32; i += 8)
        tile[i][threadIdx.x] = in[(long long)(r0 + i) * C + c0 + threadIdx.x];
    __syncthreads();
    #pragma unroll
    for (int i = threadIdx.y; i < 32; i += 8)
        out[(long long)(c0 + i) * R + r0 + threadIdx.x] = (bf16)tile[threadIdx.x][i];
}

// bf16 [R][C] -> bf16 [C][R], batched via blockIdx.z
__global__ __launch_bounds__(256) void tr_bf16(
    const bf16* __restrict__ in, bf16* __restrict__ out, int R, int C,
    long long sIn, long long sOut)
{
    __shared__ bf16 tile[32][33];
    in  += (long long)blockIdx.z * sIn;
    out += (long long)blockIdx.z * sOut;
    int r0 = blockIdx.y * 32, c0 = blockIdx.x * 32;
    #pragma unroll
    for (int i = threadIdx.y; i < 32; i += 8)
        tile[i][threadIdx.x] = in[(long long)(r0 + i) * C + c0 + threadIdx.x];
    __syncthreads();
    #pragma unroll
    for (int i = threadIdx.y; i < 32; i += 8)
        out[(long long)(c0 + i) * R + r0 + threadIdx.x] = tile[threadIdx.x][i];
}

// ---------------------------------------------------------------------------
// Generic bf16 WMMA GEMM:  C[M][N](f32) = A[M][K](bf16) x Bt[N][K](bf16)^T
// 256 threads = 8 waves (4M x 2N), each wave -> 32x64 tile (8 accumulators).
// Fused epilogue: preScale, bias, GELU, residual (+scale), f32/bf16 stores.
// flags: 1 = GELU, 2 = causal block early-exit (scores)
// ---------------------------------------------------------------------------
__global__ __launch_bounds__(256) void gemm_bf16_wmma(
    const bf16* __restrict__ A, const bf16* __restrict__ Bt,
    int N, int K,
    long long sA, long long sB, long long sC,
    const float* __restrict__ bias, const float* __restrict__ residual,
    float resScale, float preScale,
    float* __restrict__ outF, bf16* __restrict__ outH, int flags)
{
    int bm = blockIdx.y * 128;
    int bn = blockIdx.x * 128;
    if ((flags & 2) && bn > bm + 127) return;   // strictly above causal diagonal

    const bf16* Ab = A  + (long long)blockIdx.z * sA;
    const bf16* Bb = Bt + (long long)blockIdx.z * sB;
    long long zC   = (long long)blockIdx.z * sC;

    int wave = threadIdx.x >> 5;
    int lane = threadIdx.x & 31;
    int wm = (wave & 3) * 32;    // wave row offset in tile
    int wn = (wave >> 2) * 64;   // wave col offset in tile
    int lr = lane & 15;
    int hi = lane >> 4;          // lane half (0/1)

    const bf16* aRow0 = Ab + (long long)(bm + wm + lr) * K + hi * 8;
    const bf16* aRow1 = aRow0 + (long long)16 * K;
    const bf16* bRow[4];
    #pragma unroll
    for (int j = 0; j < 4; ++j)
        bRow[j] = Bb + (long long)(bn + wn + 16 * j + lr) * K + hi * 8;

    v8f zacc = {0.f, 0.f, 0.f, 0.f, 0.f, 0.f, 0.f, 0.f};
    v8f acc[2][4];
    #pragma unroll
    for (int i = 0; i < 2; ++i)
        #pragma unroll
        for (int j = 0; j < 4; ++j) acc[i][j] = zacc;

    for (int kb = 0; kb < K; kb += 32) {
        Frag16 a0, a1, b[4];
        a0.q[0] = *(const u32x4*)(aRow0 + kb);
        a0.q[1] = *(const u32x4*)(aRow0 + kb + 16);
        a1.q[0] = *(const u32x4*)(aRow1 + kb);
        a1.q[1] = *(const u32x4*)(aRow1 + kb + 16);
        #pragma unroll
        for (int j = 0; j < 4; ++j) {
            b[j].q[0] = *(const u32x4*)(bRow[j] + kb);
            b[j].q[1] = *(const u32x4*)(bRow[j] + kb + 16);
        }
        #pragma unroll
        for (int j = 0; j < 4; ++j) {
            acc[0][j] = __builtin_amdgcn_wmma_f32_16x16x32_bf16(
                false, a0.v, false, b[j].v, (short)0, acc[0][j], false, false);
            acc[1][j] = __builtin_amdgcn_wmma_f32_16x16x32_bf16(
                false, a1.v, false, b[j].v, (short)0, acc[1][j], false, false);
        }
    }

    #pragma unroll
    for (int i = 0; i < 2; ++i) {
        int rowBase = bm + wm + i * 16 + hi * 8;
        #pragma unroll
        for (int j = 0; j < 4; ++j) {
            int col = bn + wn + j * 16 + lr;
            float bcol = bias ? bias[col] : 0.0f;
            #pragma unroll
            for (int r = 0; r < 8; ++r) {
                int row = rowBase + r;
                float v = acc[i][j][r] * preScale + bcol;
                if (flags & 1)
                    v = 0.5f * v * (1.0f + erff(v * 0.70710678118654752f));
                if (residual)
                    v = residual[zC + (long long)row * N + col] + resScale * v;
                long long o = zC + (long long)row * N + col;
                if (outF) outF[o] = v;
                if (outH) outH[o] = (bf16)v;
            }
        }
    }
}

// ---------------------------------------------------------------------------
// Causal softmax over one row of scores; writes bf16 probabilities.
// grid = (T, 1, B), 256 threads, 8 cols per thread held in registers.
// Mask applied by index, so unwritten upper-triangle score tiles are ignored.
// ---------------------------------------------------------------------------
__global__ __launch_bounds__(256) void softmax_causal(
    const float* __restrict__ scores, bf16* __restrict__ probs, long long sB)
{
    int row = blockIdx.x;
    const float* s = scores + (long long)blockIdx.z * sB + (long long)row * TSEQ;
    bf16*        p = probs  + (long long)blockIdx.z * sB + (long long)row * TSEQ;
    __shared__ float red[256];

    float v[8];
    float mx = -INFINITY;
    #pragma unroll
    for (int j = 0; j < 8; ++j) {
        int col = threadIdx.x + j * 256;
        v[j] = (col <= row) ? s[col] : -INFINITY;
        mx = fmaxf(mx, v[j]);
    }
    red[threadIdx.x] = mx; __syncthreads();
    for (int st = 128; st > 0; st >>= 1) {
        if (threadIdx.x < st)
            red[threadIdx.x] = fmaxf(red[threadIdx.x], red[threadIdx.x + st]);
        __syncthreads();
    }
    mx = red[0]; __syncthreads();

    float sum = 0.0f;
    #pragma unroll
    for (int j = 0; j < 8; ++j) {
        int col = threadIdx.x + j * 256;
        float e = (col <= row) ? __expf(v[j] - mx) : 0.0f;
        v[j] = e; sum += e;
    }
    red[threadIdx.x] = sum; __syncthreads();
    for (int st = 128; st > 0; st >>= 1) {
        if (threadIdx.x < st) red[threadIdx.x] += red[threadIdx.x + st];
        __syncthreads();
    }
    float inv = 1.0f / red[0];
    #pragma unroll
    for (int j = 0; j < 8; ++j)
        p[threadIdx.x + j * 256] = (bf16)(v[j] * inv);
}

// ---------------------------------------------------------------------------
extern "C" void kernel_launch(void* const* d_in, const int* in_sizes, int n_in,
                              void* d_out, int out_size, void* d_ws, size_t ws_size,
                              hipStream_t stream)
{
    (void)in_sizes; (void)n_in; (void)out_size; (void)ws_size;
    const int*   idx  = (const int*)  d_in[0];
    const float* tok  = (const float*)d_in[1];
    const float* pos  = (const float*)d_in[2];
    const float* Wq   = (const float*)d_in[3];
    const float* Wk   = (const float*)d_in[4];
    const float* Wv   = (const float*)d_in[5];
    const float* W1   = (const float*)d_in[6];
    const float* b1   = (const float*)d_in[7];
    const float* W2   = (const float*)d_in[8];
    const float* b2   = (const float*)d_in[9];
    const float* Wout = (const float*)d_in[10];
    const float* bout = (const float*)d_in[11];
    float* logits = (float*)d_out;

    const long long BT = (long long)NBATCH * TSEQ;           // 4096
    size_t off = 0;
    auto alloc = [&](size_t bytes) -> void* {
        off = (off + 255) & ~(size_t)255;
        void* p = (char*)d_ws + off;
        off += bytes;
        return p;
    };
    float* x     = (float*)alloc(BT * DMODEL * 4);                 // embeddings f32
    bf16*  xh    = (bf16*) alloc(BT * DMODEL * 2);
    bf16*  qh    = (bf16*) alloc(BT * DMODEL * 2);
    bf16*  kh    = (bf16*) alloc(BT * DMODEL * 2);
    bf16*  vh    = (bf16*) alloc(BT * DMODEL * 2);
    bf16*  vt    = (bf16*) alloc(BT * DMODEL * 2);                 // [b][d][t]
    float* sc    = (float*)alloc((long long)NBATCH * TSEQ * TSEQ * 4);
    bf16*  attn  = (bf16*) alloc((long long)NBATCH * TSEQ * TSEQ * 2);
    float* x2    = (float*)alloc(BT * DMODEL * 4);
    bf16*  x2h   = (bf16*) alloc(BT * DMODEL * 2);
    bf16*  hh    = (bf16*) alloc(BT * 4 * DMODEL * 2);
    bf16*  x3h   = (bf16*) alloc(BT * DMODEL * 2);
    bf16*  Wqt   = (bf16*) alloc((size_t)DMODEL * DMODEL * 2);
    bf16*  Wkt   = (bf16*) alloc((size_t)DMODEL * DMODEL * 2);
    bf16*  Wvt   = (bf16*) alloc((size_t)DMODEL * DMODEL * 2);
    bf16*  W1t   = (bf16*) alloc((size_t)DMODEL * 4 * DMODEL * 2);
    bf16*  W2t   = (bf16*) alloc((size_t)4 * DMODEL * DMODEL * 2);
    bf16*  Wot   = (bf16*) alloc((size_t)DMODEL * NVOCAB * 2);

    dim3 tb(32, 8);
    // Weights -> transposed bf16 (Wt[n][k])
    wt_f32_to_bf16_t<<<dim3(DMODEL/32, DMODEL/32), tb, 0, stream>>>(Wq, Wqt, DMODEL, DMODEL);
    wt_f32_to_bf16_t<<<dim3(DMODEL/32, DMODEL/32), tb, 0, stream>>>(Wk, Wkt, DMODEL, DMODEL);
    wt_f32_to_bf16_t<<<dim3(DMODEL/32, DMODEL/32), tb, 0, stream>>>(Wv, Wvt, DMODEL, DMODEL);
    wt_f32_to_bf16_t<<<dim3(4*DMODEL/32, DMODEL/32), tb, 0, stream>>>(W1, W1t, DMODEL, 4*DMODEL);
    wt_f32_to_bf16_t<<<dim3(DMODEL/32, 4*DMODEL/32), tb, 0, stream>>>(W2, W2t, 4*DMODEL, DMODEL);
    wt_f32_to_bf16_t<<<dim3(NVOCAB/32, DMODEL/32), tb, 0, stream>>>(Wout, Wot, DMODEL, NVOCAB);

    // x = tok_emb[idx] + pos_emb
    embed_kernel<<<(int)BT, 256, 0, stream>>>(idx, tok, pos, x, xh);

    // Q, K, V projections (bf16 outputs only)
    gemm_bf16_wmma<<<dim3(DMODEL/128, BT/128, 1), 256, 0, stream>>>(
        xh, Wqt, DMODEL, DMODEL, 0, 0, 0, nullptr, nullptr, 1.f, 1.f, nullptr, qh, 0);
    gemm_bf16_wmma<<<dim3(DMODEL/128, BT/128, 1), 256, 0, stream>>>(
        xh, Wkt, DMODEL, DMODEL, 0, 0, 0, nullptr, nullptr, 1.f, 1.f, nullptr, kh, 0);
    gemm_bf16_wmma<<<dim3(DMODEL/128, BT/128, 1), 256, 0, stream>>>(
        xh, Wvt, DMODEL, DMODEL, 0, 0, 0, nullptr, nullptr, 1.f, 1.f, nullptr, vh, 0);

    // V -> V^T per batch ([t][d] -> [d][t]) for the attn@V WMMA B-operand
    tr_bf16<<<dim3(DMODEL/32, TSEQ/32, NBATCH), tb, 0, stream>>>(
        vh, vt, TSEQ, DMODEL, (long long)TSEQ * DMODEL, (long long)DMODEL * TSEQ);

    // scores = (Q K^T) / sqrt(D); upper-triangle blocks skipped (flag 2)
    gemm_bf16_wmma<<<dim3(TSEQ/128, TSEQ/128, NBATCH), 256, 0, stream>>>(
        qh, kh, TSEQ, DMODEL,
        (long long)TSEQ * DMODEL, (long long)TSEQ * DMODEL, (long long)TSEQ * TSEQ,
        nullptr, nullptr, 1.f, 0.03125f, sc, nullptr, 2);

    // causal softmax -> bf16 probabilities
    softmax_causal<<<dim3(TSEQ, 1, NBATCH), 256, 0, stream>>>(
        sc, attn, (long long)TSEQ * TSEQ);

    // x2 = x + 0.125 * (attn @ V)
    gemm_bf16_wmma<<<dim3(DMODEL/128, TSEQ/128, NBATCH), 256, 0, stream>>>(
        attn, vt, DMODEL, TSEQ,
        (long long)TSEQ * TSEQ, (long long)DMODEL * TSEQ, (long long)TSEQ * DMODEL,
        nullptr, x, 0.125f, 1.f, x2, x2h, 0);

    // h = gelu(x2 @ W1 + b1)
    gemm_bf16_wmma<<<dim3(4*DMODEL/128, BT/128, 1), 256, 0, stream>>>(
        x2h, W1t, 4*DMODEL, DMODEL, 0, 0, 0, b1, nullptr, 1.f, 1.f, nullptr, hh, 1);

    // x3 = x2 + (h @ W2 + b2)
    gemm_bf16_wmma<<<dim3(DMODEL/128, BT/128, 1), 256, 0, stream>>>(
        hh, W2t, DMODEL, 4*DMODEL, 0, 0, 0, b2, x2, 1.f, 1.f, nullptr, x3h, 0);

    // logits = x3 @ Wout + bout  -> d_out (f32)
    gemm_bf16_wmma<<<dim3(NVOCAB/128, BT/128, 1), 256, 0, stream>>>(
        x3h, Wot, NVOCAB, DMODEL, 0, 0, 0, bout, nullptr, 1.f, 1.f, logits, nullptr, 0);
}
